// AlignVector_9139690406180
// MI455X (gfx1250) — matrix operand order repaired and checked by reading, not compiled, not run on hardware
//
#include <hip/hip_runtime.h>
#include <hip/hip_bf16.h>

// out[b,q,d] = softmax(e[b,q,:]) @ sent[b,:,d]
// B=64, L2=1024, L1=1024 (K), D=768.
// Pass 1: fp32 row softmax -> bf16 probabilities in d_ws (128 MiB).
// Pass 2: bf16 WMMA GEMM (v_wmma_f32_16x16x32_bf16), fp32 accumulate,
//         double-buffered LDS (1 barrier/K-step), packed b32 transpose stores.

typedef __attribute__((ext_vector_type(16))) __bf16 v16bf;
typedef __attribute__((ext_vector_type(8)))  float  v8f;

#define BATCH 64
#define L2DIM 1024
#define KDIM  1024
#define DDIM  768
#define BM 128
#define BN 128
#define BK 32
#define NK (KDIM / BK)
#define LDT 48   // padded LDS row stride in bf16 elems (96 B: keeps 16B chunks aligned)

// ---------------- Pass 1: softmax over last axis, emit bf16 ----------------
__global__ __launch_bounds__(256)
void AlignVector_softmax_bf16(const float* __restrict__ e, __bf16* __restrict__ es) {
    const int q   = blockIdx.x;
    const int b   = blockIdx.y;
    const int tid = threadIdx.x;

    const float* row = e + ((size_t)b * L2DIM + q) * KDIM;
    float4 v = ((const float4*)row)[tid];           // 256 threads * 4 = 1024

    __shared__ float red[256];

    float m = fmaxf(fmaxf(v.x, v.y), fmaxf(v.z, v.w));
    red[tid] = m;
    __syncthreads();
    for (int s = 128; s > 0; s >>= 1) {
        if (tid < s) red[tid] = fmaxf(red[tid], red[tid + s]);
        __syncthreads();
    }
    m = red[0];
    __syncthreads();

    float e0 = __expf(v.x - m);
    float e1 = __expf(v.y - m);
    float e2 = __expf(v.z - m);
    float e3 = __expf(v.w - m);
    red[tid] = e0 + e1 + e2 + e3;
    __syncthreads();
    for (int s = 128; s > 0; s >>= 1) {
        if (tid < s) red[tid] += red[tid + s];
        __syncthreads();
    }
    const float inv = 1.0f / red[0];

    union { __bf16 h[4]; uint2 u; } pk;
    pk.h[0] = (__bf16)(e0 * inv);
    pk.h[1] = (__bf16)(e1 * inv);
    pk.h[2] = (__bf16)(e2 * inv);
    pk.h[3] = (__bf16)(e3 * inv);
    ((uint2*)(es + ((size_t)b * L2DIM + q) * KDIM))[tid] = pk.u;
}

// ---------------- Pass 2: out = es @ sent via bf16 WMMA ----------------
struct Stage {
    uint4  a[2];            // A: 2 x 8 bf16 chunks
    float4 b0[2], b1[2];    // B: 2 x (two adjacent K rows, 4 d-cols each)
};

__device__ __forceinline__ unsigned pack_bf16x2(float lo, float hi) {
    union { __bf16 h[2]; unsigned u; } p;
    p.h[0] = (__bf16)lo;
    p.h[1] = (__bf16)hi;
    return p.u;
}

__device__ __forceinline__ void load_stage(Stage& s, const __bf16* esB,
                                           const float* snB, int kk, int tid) {
    #pragma unroll
    for (int j = 0; j < 2; ++j) {           // A tile: 128x32 bf16, 512 chunks of 8
        int i  = tid + 256 * j;
        int r  = i >> 2;                    // q row 0..127
        int c8 = (i & 3) << 3;              // k col 0,8,16,24
        s.a[j] = *(const uint4*)(esB + (size_t)r * KDIM + kk + c8);
    }
    #pragma unroll
    for (int j = 0; j < 2; ++j) {           // B tile: 32(k) x 128(d), 512 items
        int i  = tid + 256 * j;
        int r2 = i >> 5;                    // k-pair 0..15
        int c  = (i & 31) << 2;             // d col 0..124
        s.b0[j] = *(const float4*)(snB + (size_t)(kk + 2 * r2 + 0) * DDIM + c);
        s.b1[j] = *(const float4*)(snB + (size_t)(kk + 2 * r2 + 1) * DDIM + c);
    }
}

__device__ __forceinline__ void store_stage(const Stage& s, __bf16* As,
                                            __bf16* Bs, int tid) {
    #pragma unroll
    for (int j = 0; j < 2; ++j) {
        int i  = tid + 256 * j;
        int r  = i >> 2;
        int c8 = (i & 3) << 3;
        *(uint4*)(&As[r * LDT + c8]) = s.a[j];
    }
    #pragma unroll
    for (int j = 0; j < 2; ++j) {
        int i  = tid + 256 * j;
        int r2 = i >> 5;
        int c  = (i & 31) << 2;
        // Bs[d][k]: pack (k, k+1) bf16 pair -> one dword store per d column
        *(unsigned*)(&Bs[(c + 0) * LDT + 2 * r2]) = pack_bf16x2(s.b0[j].x, s.b1[j].x);
        *(unsigned*)(&Bs[(c + 1) * LDT + 2 * r2]) = pack_bf16x2(s.b0[j].y, s.b1[j].y);
        *(unsigned*)(&Bs[(c + 2) * LDT + 2 * r2]) = pack_bf16x2(s.b0[j].z, s.b1[j].z);
        *(unsigned*)(&Bs[(c + 3) * LDT + 2 * r2]) = pack_bf16x2(s.b0[j].w, s.b1[j].w);
    }
}

__global__ __launch_bounds__(256)
void AlignVector_wmma_pool(const __bf16* __restrict__ es,
                           const float*  __restrict__ sent,
                           float*        __restrict__ out) {
    const int dt  = blockIdx.x;           // D tile: 0..5
    const int qt  = blockIdx.y;           // L2 tile: 0..7
    const int b   = blockIdx.z;           // batch
    const int tid = threadIdx.x;
    const int lane  = tid & 31;
    const int wave  = tid >> 5;           // 0..7
    const int wm    = wave & 3;           // wave M offset: wm*32
    const int wn    = wave >> 2;          // wave N offset: wn*64
    const int lmod  = lane & 15;
    const int lhalf = lane >> 4;          // 0: lanes 0-15, 1: lanes 16-31

    __shared__ __align__(32) __bf16 As[2][BM * LDT];  // es tile: [q_row][k]
    __shared__ __align__(32) __bf16 Bs[2][BN * LDT];  // sent tile transposed: [d_col][k]

    const int q0 = qt * BM;
    const int d0 = dt * BN;

    v8f acc[2][4];
    #pragma unroll
    for (int i = 0; i < 2; ++i)
        #pragma unroll
        for (int j = 0; j < 4; ++j) acc[i][j] = (v8f){0,0,0,0,0,0,0,0};

    const __bf16* esB = es   + ((size_t)b * L2DIM + q0) * KDIM;
    const float*  snB = sent + (size_t)b * KDIM * DDIM + d0;

    Stage st;
    load_stage(st, esB, snB, 0, tid);
    store_stage(st, As[0], Bs[0], tid);
    __syncthreads();

    for (int k = 0; k < NK; ++k) {
        const int cur = k & 1;
        const bool more = (k + 1) < NK;
        if (more) load_stage(st, esB, snB, (k + 1) * BK, tid);

        // ---- fragments per ISA VGPR layouts ----
        union Frag { v16bf v; uint4 u[2]; };
        Frag a[2], bf[4];
        // A 16x32: lanes 0-15 hold K [0,8)+[16,24); lanes 16-31 hold K [8,16)+[24,32)
        #pragma unroll
        for (int mi = 0; mi < 2; ++mi) {
            const __bf16* p = &As[cur][(wm * 32 + mi * 16 + lmod) * LDT];
            a[mi].u[0] = *(const uint4*)(p + lhalf * 8);
            a[mi].u[1] = *(const uint4*)(p + 16 + lhalf * 8);
        }
        // B 32x16: lane n holds column n; lanes 0-15 K [0,8)+[16,24); lanes 16-31 K [8,16)+[24,32)
        #pragma unroll
        for (int ni = 0; ni < 4; ++ni) {
            const __bf16* p = &Bs[cur][(wn * 64 + ni * 16 + lmod) * LDT + lhalf * 8];
            bf[ni].u[0] = *(const uint4*)p;
            bf[ni].u[1] = *(const uint4*)(p + 16);
        }

        #pragma unroll
        for (int mi = 0; mi < 2; ++mi)
            #pragma unroll
            for (int ni = 0; ni < 4; ++ni)
                acc[mi][ni] = __builtin_amdgcn_wmma_f32_16x16x32_bf16(
                    /*neg_a=*/false, a[mi].v, /*neg_b=*/false, bf[ni].v,
                    /*c_mod=*/(short)0, acc[mi][ni],
                    /*reuse_a=*/false, /*reuse_b=*/false);

        if (more) {
            store_stage(st, As[1 - cur], Bs[1 - cur], tid);
            __syncthreads();
        }
    }

    // ---- store: C/D layout: VGPR j -> M=j (lanes 0-15) / M=8+j (lanes 16-31), N=lane%16
    #pragma unroll
    for (int mi = 0; mi < 2; ++mi) {
        #pragma unroll
        for (int ni = 0; ni < 4; ++ni) {
            const int rbase = q0 + wm * 32 + mi * 16 + lhalf * 8;
            const int col   = d0 + wn * 64 + ni * 16 + lmod;
            #pragma unroll
            for (int j = 0; j < 8; ++j)
                out[((size_t)b * L2DIM + rbase + j) * DDIM + col] = acc[mi][ni][j];
        }
    }
}

extern "C" void kernel_launch(void* const* d_in, const int* in_sizes, int n_in,
                              void* d_out, int out_size, void* d_ws, size_t ws_size,
                              hipStream_t stream) {
    const float* sent = (const float*)d_in[0];   // [64,1024,768] fp32
    const float* e    = (const float*)d_in[1];   // [64,1024,1024] fp32
    float* out        = (float*)d_out;           // [64,1024,768] fp32
    __bf16* es        = (__bf16*)d_ws;           // 64*1024*1024 bf16 = 128 MiB scratch

    dim3 gs(L2DIM, BATCH);
    AlignVector_softmax_bf16<<<gs, 256, 0, stream>>>(e, es);

    dim3 gg(DDIM / BN, L2DIM / BM, BATCH);
    AlignVector_wmma_pool<<<gg, 256, 0, stream>>>(es, sent, out);
}
